// Block1_75651553952212
// MI455X (gfx1250) — matrix operand order repaired and checked by reading, not compiled, not run on hardware
//
#include <hip/hip_runtime.h>
#include <hip/hip_bf16.h>
#include <math.h>

// ---------------------------------------------------------------------------
// ViT encoder block (LN -> QKV -> masked attention -> proj+res -> LN -> MLP+res)
// for MI455X / gfx1250: all GEMMs via v_wmma_f32_16x16x32_f16 (wave32).
// GEMM waves compute 32x64 output tiles (2 A-frags x 4 B-frags -> 8 WMMA/K-step)
// to maximize WMMA density per vector load; weights stream through L2 with
// global_prefetch_b8 lookahead.
// ---------------------------------------------------------------------------

typedef _Float16 v16h __attribute__((ext_vector_type(16)));
typedef _Float16 h2   __attribute__((ext_vector_type(2)));
typedef float    v8f  __attribute__((ext_vector_type(8)));

#define B_   32
#define N_   577
#define C_   768
#define H_   12
#define D_   64
#define HID_ 3072
#define M_   (B_ * N_)      // 18464 = 577*32 : multiple of 32
#define NT_  37             // ceil(577/16) key tiles
#define NPAD 608            // 19*32, padded key length for K=32 WMMA steps
#define SSTR 616            // f32 score row stride (pad vs bank conflicts)
#define PSTR 624            // f16 prob row stride (even -> half2 aligned)

// K-index striping within a fragment (per CDNA5 ISA, 16-bit ops)
__device__ __forceinline__ int k0A(int i, int hf) {
    return ((i < 4) ? (2 * i) : (16 + 2 * (i - 4))) + 8 * hf;   // A: 16x32, lane=row
}
__device__ __forceinline__ int k0B(int i, int hf) {
    return ((i < 4) ? (2 * i) : (16 + 2 * (i - 4))) + 8 * hf;   // B: 32x16, lane=col
}

__device__ __forceinline__ v8f wmma16(v16h a, v16h b, v8f c) {
    return __builtin_amdgcn_wmma_f32_16x16x32_f16(
        /*neg_a=*/false, a, /*neg_b=*/false, b,
        /*c_mod=*/(short)0, c, /*reuse_a=*/false, /*reuse_b=*/false);
}

__device__ __forceinline__ v16h load_fragA(const _Float16* __restrict__ p, int hf) {
    v16h f;
#pragma unroll
    for (int i = 0; i < 8; ++i) {
        h2 q = *(const h2*)(p + k0A(i, hf));
        f[2 * i] = q[0]; f[2 * i + 1] = q[1];
    }
    return f;
}
__device__ __forceinline__ v16h load_fragB(const _Float16* __restrict__ p, int hf) {
    v16h f;
#pragma unroll
    for (int i = 0; i < 8; ++i) {
        h2 q = *(const h2*)(p + k0B(i, hf));
        f[2 * i] = q[0]; f[2 * i + 1] = q[1];
    }
    return f;
}

// ---------------------------------------------------------------------------
// fp32 -> f16 weight conversion
// ---------------------------------------------------------------------------
__global__ void cvt_f16_kernel(const float* __restrict__ in,
                               _Float16* __restrict__ out, int n) {
    int i = blockIdx.x * blockDim.x + threadIdx.x;
    if (i < n) out[i] = (_Float16)in[i];
}

// ---------------------------------------------------------------------------
// LayerNorm over C=768, f32 in -> f16 out
// ---------------------------------------------------------------------------
__global__ __launch_bounds__(128)
void ln_kernel(const float* __restrict__ x, const float* __restrict__ g,
               const float* __restrict__ b, _Float16* __restrict__ out) {
    const int row = blockIdx.x;
    const float* xr = x + (size_t)row * C_;
    __shared__ float s1[128], s2[128];
    const int t = threadIdx.x;
    float sum = 0.f, sq = 0.f;
    for (int c = t; c < C_; c += 128) { float v = xr[c]; sum += v; sq += v * v; }
    s1[t] = sum; s2[t] = sq;
    __syncthreads();
    for (int off = 64; off > 0; off >>= 1) {
        if (t < off) { s1[t] += s1[t + off]; s2[t] += s2[t + off]; }
        __syncthreads();
    }
    const float mu  = s1[0] * (1.f / C_);
    const float var = s2[0] * (1.f / C_) - mu * mu;
    const float rs  = rsqrtf(var + 1e-5f);
    _Float16* orow = out + (size_t)row * C_;
    for (int c = t; c < C_; c += 128)
        orow[c] = (_Float16)((xr[c] - mu) * rs * g[c] + b[c]);
}

// ---------------------------------------------------------------------------
// Generic WMMA GEMM: out[m][n] = sum_k A[m][k] * W[n][k]  (+ epilogue)
//   MODE 0: store f16                        (QKV)
//   MODE 1: +bias +residual, store f32       (proj)
//   MODE 2: +bias, exact GELU, store f16     (fc1)
//   MODE 3: +bias +residual, store f32       (fc2 -> d_out)
// Block 256 thr = 8 waves (4 Mtiles x 2 Ngroups); each wave: 32 rows x 64 cols
// (8 WMMAs per 32-wide K step, A-frags reused across 4 N-tiles, B-frags across
// 2 M-subtiles).
// ---------------------------------------------------------------------------
template <int MODE>
__global__ __launch_bounds__(256)
void gemm_wmma(const _Float16* __restrict__ A, int lda,
               const _Float16* __restrict__ W, int ldw,
               const float* __restrict__ bias,
               const float* __restrict__ resid,
               float* __restrict__ outF, _Float16* __restrict__ outH,
               int ldo, int M, int K) {
    const int lane  = threadIdx.x & 31;
    const int wave  = threadIdx.x >> 5;
    const int mBase = blockIdx.y * 128 + (wave >> 1) * 32;
    if (mBase >= M) return;                         // wave-uniform guard; M%32==0
    const int nBase = blockIdx.x * 128 + (wave & 1) * 64;
    const int r  = lane & 15;
    const int hf = lane >> 4;

    v8f zero = {};
    v8f acc[2][4];
#pragma unroll
    for (int s = 0; s < 2; ++s)
#pragma unroll
        for (int t = 0; t < 4; ++t) acc[s][t] = zero;

    const _Float16* aRow0 = A + (size_t)(mBase + r) * lda;
    const _Float16* aRow1 = aRow0 + (size_t)16 * lda;
    for (int kb = 0; kb < K; kb += 32) {
        // prefetch next K block of the streaming operands (global_prefetch_b8)
        if (kb + 32 < K) {
            __builtin_prefetch(aRow0 + kb + 32, 0, 1);
            __builtin_prefetch(W + (size_t)(nBase + r) * ldw + kb + 32, 0, 1);
        }
        v16h af0 = load_fragA(aRow0 + kb, hf);
        v16h af1 = load_fragA(aRow1 + kb, hf);
#pragma unroll
        for (int t = 0; t < 4; ++t) {
            const _Float16* wRow = W + (size_t)(nBase + t * 16 + r) * ldw;
            v16h bf = load_fragB(wRow + kb, hf);
            acc[0][t] = wmma16(af0, bf, acc[0][t]);
            acc[1][t] = wmma16(af1, bf, acc[1][t]);
        }
    }

#pragma unroll
    for (int t = 0; t < 4; ++t) {
        const int col = nBase + t * 16 + r;
        float bv = 0.f;
        if (MODE != 0) bv = bias[col];
#pragma unroll
        for (int s = 0; s < 2; ++s) {
#pragma unroll
            for (int j = 0; j < 8; ++j) {
                const int row = mBase + s * 16 + j + 8 * hf;
                float v = acc[s][t][j] + bv;
                const size_t idx = (size_t)row * ldo + col;
                if (MODE == 0) {
                    outH[idx] = (_Float16)v;
                } else if (MODE == 1 || MODE == 3) {
                    outF[idx] = v + resid[idx];
                } else {  // MODE 2: exact GELU
                    float gl = 0.5f * v * (1.f + erff(v * 0.70710678118654752f));
                    outH[idx] = (_Float16)gl;
                }
            }
        }
    }
}

// ---------------------------------------------------------------------------
// Attention: one wave per (b, h, 16-row tile). Scores->LDS, softmax + task
// mask (post-softmax, 3x3 diag), P@V via WMMA.  qkv layout [B*N, 3C] f16.
// ---------------------------------------------------------------------------
__global__ __launch_bounds__(32)
void attn_kernel(const _Float16* __restrict__ qkv, _Float16* __restrict__ o) {
    __shared__ float    S[16 * SSTR];   // scores (f32)
    __shared__ _Float16 P[16 * PSTR];   // probabilities (f16), zero-padded

    const int mt = blockIdx.x, h = blockIdx.y, b = blockIdx.z;
    const int lane = threadIdx.x;
    const int r = lane & 15, hf = lane >> 4;
    const size_t baseRow = (size_t)b * N_;

    // Q fragment (rows of this tile), clamped for the padded tail rows
    int qm = mt * 16 + r; if (qm > N_ - 1) qm = N_ - 1;
    const _Float16* qrow = qkv + (baseRow + qm) * (3 * C_) + (size_t)h * D_;
    v16h qa0 = load_fragA(qrow, hf);
    v16h qa1 = load_fragA(qrow + 32, hf);

    // S = (Q K^T) * scale, tile by tile into LDS
    for (int j = 0; j < NT_; ++j) {
        int kn = j * 16 + r; if (kn > N_ - 1) kn = N_ - 1;
        const _Float16* krow = qkv + (baseRow + kn) * (3 * C_) + C_ + (size_t)h * D_;
        v16h kb0 = load_fragB(krow, hf);
        v16h kb1 = load_fragB(krow + 32, hf);
        v8f acc = {};
        acc = wmma16(qa0, kb0, acc);
        acc = wmma16(qa1, kb1, acc);
        const int col = j * 16 + r;
#pragma unroll
        for (int jj = 0; jj < 8; ++jj) {
            float s = acc[jj] * 0.125f;                 // D^-0.5 = 1/8
            if (col >= N_) s = -1e30f;                  // padded keys
            S[(jj + 8 * hf) * SSTR + col] = s;
        }
    }
    __syncthreads();

    // Softmax along keys; 2 lanes per row, combined via shfl_xor(16)
    {
        float mx = -1e30f;
        for (int c = hf; c < N_; c += 2) mx = fmaxf(mx, S[r * SSTR + c]);
        mx = fmaxf(mx, __shfl_xor(mx, 16, 32));
        float sm = 0.f;
        for (int c = hf; c < N_; c += 2) sm += expf(S[r * SSTR + c] - mx);
        sm += __shfl_xor(sm, 16, 32);
        const float inv = 1.f / sm;
        const int gi = mt * 16 + r;
        for (int c = hf; c < NPAD; c += 2) {
            float pv = 0.f;
            if (c < N_) {
                pv = expf(S[r * SSTR + c] - mx) * inv;
                if (gi < 3 && c < 3 && c != gi) pv = 0.f;   // post-softmax task mask
            }
            P[r * PSTR + c] = (_Float16)pv;
        }
    }
    __syncthreads();

    // O = P @ V : 4 d-tiles of 16, K loop over 608 padded keys
#pragma unroll
    for (int dt = 0; dt < 4; ++dt) {
        v8f acc = {};
        const int d = dt * 16 + r;
        for (int kk = 0; kk < NPAD / 32; ++kk) {
            v16h pa = load_fragA(P + r * PSTR + kk * 32, hf);
            v16h vb;
#pragma unroll
            for (int i = 0; i < 8; ++i) {
                int kr0 = kk * 32 + k0B(i, hf);
                int kr1 = kr0 + 1;
                if (kr0 > N_ - 1) kr0 = N_ - 1;             // P==0 there anyway
                if (kr1 > N_ - 1) kr1 = N_ - 1;
                vb[2 * i]     = qkv[(baseRow + kr0) * (3 * C_) + 2 * C_ + (size_t)h * D_ + d];
                vb[2 * i + 1] = qkv[(baseRow + kr1) * (3 * C_) + 2 * C_ + (size_t)h * D_ + d];
            }
            acc = wmma16(pa, vb, acc);
        }
#pragma unroll
        for (int jj = 0; jj < 8; ++jj) {
            const int m = mt * 16 + jj + 8 * hf;
            if (m < N_)
                o[(baseRow + m) * C_ + (size_t)h * D_ + d] = (_Float16)acc[jj];
        }
    }
}

// ---------------------------------------------------------------------------
// Host-side orchestration
// ---------------------------------------------------------------------------
extern "C" void kernel_launch(void* const* d_in, const int* in_sizes, int n_in,
                              void* d_out, int out_size, void* d_ws, size_t ws_size,
                              hipStream_t stream) {
    (void)in_sizes; (void)n_in; (void)out_size; (void)ws_size;

    const float* x      = (const float*)d_in[0];
    const float* w_qkv  = (const float*)d_in[1];
    const float* w_proj = (const float*)d_in[2];
    const float* b_proj = (const float*)d_in[3];
    const float* ln1_g  = (const float*)d_in[4];
    const float* ln1_b  = (const float*)d_in[5];
    const float* ln2_g  = (const float*)d_in[6];
    const float* ln2_b  = (const float*)d_in[7];
    const float* w_fc1  = (const float*)d_in[8];
    const float* b_fc1  = (const float*)d_in[9];
    const float* w_fc2  = (const float*)d_in[10];
    const float* b_fc2  = (const float*)d_in[11];
    float* out = (float*)d_out;

    char* ws = (char*)d_ws;
    size_t off = 0;
    auto take = [&](size_t bytes) -> void* {
        void* p = ws + off;
        off += (bytes + 255) & ~(size_t)255;
        return p;
    };
    _Float16* wq16  = (_Float16*)take((size_t)3 * C_ * C_ * 2);
    _Float16* wp16  = (_Float16*)take((size_t)C_ * C_ * 2);
    _Float16* wf1   = (_Float16*)take((size_t)HID_ * C_ * 2);
    _Float16* wf2   = (_Float16*)take((size_t)C_ * HID_ * 2);
    _Float16* h16   = (_Float16*)take((size_t)M_ * C_ * 2);
    _Float16* qkv16 = (_Float16*)take((size_t)M_ * 3 * C_ * 2);
    _Float16* o16   = (_Float16*)take((size_t)M_ * C_ * 2);
    float*    xa    = (float*)   take((size_t)M_ * C_ * 4);
    _Float16* hm16  = (_Float16*)take((size_t)M_ * HID_ * 2);

    // weight conversion to f16
    auto cvt = [&](const float* src, _Float16* dst, int n) {
        cvt_f16_kernel<<<(n + 255) / 256, 256, 0, stream>>>(src, dst, n);
    };
    cvt(w_qkv,  wq16, 3 * C_ * C_);
    cvt(w_proj, wp16, C_ * C_);
    cvt(w_fc1,  wf1,  HID_ * C_);
    cvt(w_fc2,  wf2,  C_ * HID_);

    const dim3 blk256(256);
    const int gy = (M_ + 127) / 128;

    // LN1
    ln_kernel<<<M_, 128, 0, stream>>>(x, ln1_g, ln1_b, h16);
    // QKV GEMM -> f16 [M, 3C]
    gemm_wmma<0><<<dim3(3 * C_ / 128, gy), blk256, 0, stream>>>(
        h16, C_, wq16, C_, nullptr, nullptr, nullptr, qkv16, 3 * C_, M_, C_);
    // masked multi-head attention -> o16 [M, C]
    attn_kernel<<<dim3(NT_, H_, B_), dim3(32), 0, stream>>>(qkv16, o16);
    // proj + bias + residual -> xa (f32)
    gemm_wmma<1><<<dim3(C_ / 128, gy), blk256, 0, stream>>>(
        o16, C_, wp16, C_, b_proj, x, xa, nullptr, C_, M_, C_);
    // LN2
    ln_kernel<<<M_, 128, 0, stream>>>(xa, ln2_g, ln2_b, h16);
    // fc1 + bias + GELU -> hm16 (f16)
    gemm_wmma<2><<<dim3(HID_ / 128, gy), blk256, 0, stream>>>(
        h16, C_, wf1, C_, b_fc1, nullptr, nullptr, hm16, HID_, M_, C_);
    // fc2 + bias + residual -> d_out (f32)
    gemm_wmma<3><<<dim3(C_ / 128, gy), blk256, 0, stream>>>(
        hm16, HID_, wf2, HID_, b_fc2, xa, out, nullptr, C_, M_, HID_);
}
